// EvolveGCNModel_25451976196928
// MI455X (gfx1250) — matrix operand order repaired and checked by reading, not compiled, not run on hardware
//
#include <hip/hip_runtime.h>
#include <hip/hip_bf16.h>
#include <stdint.h>

typedef __attribute__((ext_vector_type(16))) __bf16 v16bf;
typedef __attribute__((ext_vector_type(8)))  float  v8f;

#define FDIM 128
#define TGT  8

__device__ __forceinline__ float sigmoidf_(float x) { return 1.0f / (1.0f + __expf(-x)); }
__device__ __forceinline__ float tanh_fast(float x) { return 1.0f - 2.0f / (__expf(2.0f * x) + 1.0f); }

// ---------------------------------------------------------------------------
// Kernel 1: GRU weight evolution in fp32, writing W directly in WMMA-B lane
// layout (bf16 hi + bf16 residual lo) for the big GEMM.
// Grid: 64 x 256 = 16384 threads = one per (i,j) element of W [128,128].
// ---------------------------------------------------------------------------
__global__ void gates_kernel(const float* __restrict__ W0, const float* __restrict__ Wih,
                             const float* __restrict__ Whh, const float* __restrict__ bih,
                             const float* __restrict__ bhh,
                             __bf16* __restrict__ Wb_hi, __bf16* __restrict__ Wb_lo) {
    int tid = blockIdx.x * blockDim.x + threadIdx.x;   // 0..16383
    int i = tid >> 7;          // row of W0 (k index of B)
    int j = tid & 127;         // col (n index of B)
    const float* a  = W0 + i * FDIM;
    const float* r0 = Wih + (size_t)j * FDIM;
    const float* r1 = Wih + (size_t)(j + 128) * FDIM;
    const float* r2 = Wih + (size_t)(j + 256) * FDIM;
    const float* h0 = Whh + (size_t)j * FDIM;
    const float* h1 = Whh + (size_t)(j + 128) * FDIM;
    const float* h2 = Whh + (size_t)(j + 256) * FDIM;
    float s0 = 0.f, s1 = 0.f, s2 = 0.f, s3 = 0.f, s4 = 0.f, s5 = 0.f;
#pragma unroll 4
    for (int k = 0; k < FDIM; ++k) {
        float av = a[k];
        s0 = fmaf(av, r0[k], s0); s1 = fmaf(av, r1[k], s1); s2 = fmaf(av, r2[k], s2);
        s3 = fmaf(av, h0[k], s3); s4 = fmaf(av, h1[k], s4); s5 = fmaf(av, h2[k], s5);
    }
    float ir = s0 + bih[j], iz = s1 + bih[j + 128], inn = s2 + bih[j + 256];
    float hr = s3 + bhh[j], hz = s4 + bhh[j + 128], hn  = s5 + bhh[j + 256];
    float r = sigmoidf_(ir + hr);
    float z = sigmoidf_(iz + hz);
    float n = tanh_fast(inn + r * hn);
    float w = (1.0f - z) * n + z * a[j];            // new W[i][j]
    __bf16 whi = (__bf16)w;
    __bf16 wlo = (__bf16)(w - (float)whi);
    // B-operand layout for v_wmma_f32_16x16x32_bf16 (B is 32x16 per tile):
    // tile (kt,nt); within tile: lane = n + 16*((k_in_tile/8)%2); elem = 8*(k_in_tile/16) + k%8
    int kt = i >> 5, kk = i & 31, nt = j >> 4, nn = j & 15;
    int lane = nn + (((kk >> 3) & 1) << 4);
    int idx  = (((kt * 8 + nt) * 32) + lane) * 16 + ((kk >> 4) << 3) + (kk & 7);
    Wb_hi[idx] = whi;
    Wb_lo[idx] = wlo;
}

// ---------------------------------------------------------------------------
// Degree / normalization kernels
// ---------------------------------------------------------------------------
__global__ void deg_init_kernel(float* __restrict__ deg, int N) {
    int n = blockIdx.x * blockDim.x + threadIdx.x;
    if (n < N) deg[n] = 1.0f;                       // self-loop weight
}

__global__ void deg_accum_kernel(const int* __restrict__ col, const float* __restrict__ ew,
                                 float* __restrict__ deg, int E) {
    int e = blockIdx.x * blockDim.x + threadIdx.x;
    if (e < E) unsafeAtomicAdd(&deg[col[e]], ew[e]);
}

__global__ void dinv_kernel(const float* __restrict__ deg, float* __restrict__ dinv, int N) {
    int n = blockIdx.x * blockDim.x + threadIdx.x;
    if (n < N) {
        float d = deg[n];
        dinv[n] = (d > 0.f) ? rsqrtf(d) : 0.f;
    }
}

// ---------------------------------------------------------------------------
// Kernel: xw = x @ W with bf16 split WMMA (hi*hi + hi*lo + lo*hi ~ fp32).
// One wave per 16-row tile; W (B operand) preloaded into LDS in lane layout.
// ---------------------------------------------------------------------------
__global__ void __launch_bounds__(128)
gemm_xw_kernel(const float* __restrict__ x, const __bf16* __restrict__ Wb_hi,
               const __bf16* __restrict__ Wb_lo, float* __restrict__ xw,
               int N, int tiles) {
    __shared__ uint32_t sB[16384];                   // 32KB hi + 32KB lo
    const uint32_t* gh = (const uint32_t*)Wb_hi;
    const uint32_t* gl = (const uint32_t*)Wb_lo;
    for (int idx = threadIdx.x; idx < 8192; idx += 128) {
        sB[idx]        = gh[idx];
        sB[8192 + idx] = gl[idx];
    }
    __syncthreads();

    int tile = blockIdx.x * 4 + (threadIdx.x >> 5);
    if (tile >= tiles) return;
    int ln = threadIdx.x & 31;

    int mrow  = tile * 16 + (ln & 15);
    int mload = mrow < N ? mrow : N - 1;
    int c0    = (ln < 16) ? 0 : 8;                   // A lane K-subset base

    v8f acc[8];
#pragma unroll
    for (int q = 0; q < 8; ++q) acc[q] = (v8f){0.f, 0.f, 0.f, 0.f, 0.f, 0.f, 0.f, 0.f};

#pragma unroll
    for (int kt = 0; kt < 4; ++kt) {
        const float* p = x + (size_t)mload * FDIM + kt * 32 + c0;
        float4 f0 = *(const float4*)(p);
        float4 f1 = *(const float4*)(p + 4);
        float4 f2 = *(const float4*)(p + 16);
        float4 f3 = *(const float4*)(p + 20);
        float xs[16] = {f0.x, f0.y, f0.z, f0.w, f1.x, f1.y, f1.z, f1.w,
                        f2.x, f2.y, f2.z, f2.w, f3.x, f3.y, f3.z, f3.w};
        v16bf ahi, alo;
#pragma unroll
        for (int q = 0; q < 16; ++q) {
            __bf16 hi = (__bf16)xs[q];
            ahi[q] = hi;
            alo[q] = (__bf16)(xs[q] - (float)hi);
        }
#pragma unroll
        for (int nt = 0; nt < 8; ++nt) {
            int toff = ((kt * 8 + nt) * 32 + ln) << 3;     // dword offset
            const v16bf bhi = *(const v16bf*)(sB + toff);
            const v16bf blo = *(const v16bf*)(sB + 8192 + toff);
            acc[nt] = __builtin_amdgcn_wmma_f32_16x16x32_bf16(
                false, ahi, false, bhi, (short)0, acc[nt], false, false);
            acc[nt] = __builtin_amdgcn_wmma_f32_16x16x32_bf16(
                false, ahi, false, blo, (short)0, acc[nt], false, false);
            acc[nt] = __builtin_amdgcn_wmma_f32_16x16x32_bf16(
                false, alo, false, bhi, (short)0, acc[nt], false, false);
        }
    }

    // C layout: VGPR r -> row m0 + r + 8*(lane>=16); col = n0 + (lane&15)
    int rbase = tile * 16 + ((ln >> 4) << 3);
#pragma unroll
    for (int nt = 0; nt < 8; ++nt) {
#pragma unroll
        for (int rr = 0; rr < 8; ++rr) {
            int row = rbase + rr;
            if (row < N) xw[(size_t)row * FDIM + nt * 16 + (ln & 15)] = acc[nt][rr];
        }
    }
}

// ---------------------------------------------------------------------------
// h init with self-loop term: h[n][f] = dinv[n]^2 * xw[n][f]
// ---------------------------------------------------------------------------
__global__ void hinit_kernel(const float* __restrict__ xw, const float* __restrict__ dinv,
                             float* __restrict__ h, int total) {
    int idx = blockIdx.x * blockDim.x + threadIdx.x;
    if (idx < total) {
        int n = idx >> 7;
        float di = dinv[n];
        h[idx] = di * di * xw[idx];
    }
}

// ---------------------------------------------------------------------------
// Edge scatter: one wave per edge; lanes cover 128 feats as float4.
// xw/h are L2-resident (25.6MB each vs 192MB L2).
// ---------------------------------------------------------------------------
__global__ void scatter_kernel(const int* __restrict__ row, const int* __restrict__ col,
                               const float* __restrict__ ew, const float* __restrict__ dinv,
                               const float* __restrict__ xw, float* __restrict__ h,
                               int E, int nwaves) {
    int wave = (blockIdx.x * blockDim.x + threadIdx.x) >> 5;
    int ln   = threadIdx.x & 31;
    for (int e = wave; e < E; e += nwaves) {
        int r = row[e], c = col[e];
        float w = dinv[r] * ew[e] * dinv[c];
        const float4 v = *(const float4*)(xw + (size_t)r * FDIM + ln * 4);
        float* hp = h + (size_t)c * FDIM + ln * 4;
        unsafeAtomicAdd(hp + 0, w * v.x);
        unsafeAtomicAdd(hp + 1, w * v.y);
        unsafeAtomicAdd(hp + 2, w * v.z);
        unsafeAtomicAdd(hp + 3, w * v.w);
    }
}

// ---------------------------------------------------------------------------
// out = relu(h) @ Wlin^T + blin, T=8. One wave per node, shuffle reduction.
// ---------------------------------------------------------------------------
__global__ void final_kernel(const float* __restrict__ h, const float* __restrict__ Wlin,
                             const float* __restrict__ blin, float* __restrict__ out, int N) {
    __shared__ float sW[TGT * FDIM];
    __shared__ float sb[TGT];
    for (int idx = threadIdx.x; idx < TGT * FDIM; idx += blockDim.x) sW[idx] = Wlin[idx];
    if (threadIdx.x < TGT) sb[threadIdx.x] = blin[threadIdx.x];
    __syncthreads();

    int node = blockIdx.x * 8 + (threadIdx.x >> 5);
    if (node >= N) return;
    int ln  = threadIdx.x & 31;
    int t   = ln & 7;
    int seg = ln >> 3;
    const float* hr = h  + (size_t)node * FDIM + seg * 32;
    const float* wl = sW + t * FDIM + seg * 32;
    float s = 0.f;
#pragma unroll
    for (int k = 0; k < 32; k += 4) {
        float4 hv = *(const float4*)(hr + k);
        float4 wv = *(const float4*)(wl + k);
        s += fmaxf(hv.x, 0.f) * wv.x + fmaxf(hv.y, 0.f) * wv.y +
             fmaxf(hv.z, 0.f) * wv.z + fmaxf(hv.w, 0.f) * wv.w;
    }
    s += __shfl_xor(s, 8, 32);
    s += __shfl_xor(s, 16, 32);
    if (ln < 8) out[(size_t)node * TGT + t] = s + sb[t];
}

// ---------------------------------------------------------------------------
extern "C" void kernel_launch(void* const* d_in, const int* in_sizes, int n_in,
                              void* d_out, int out_size, void* d_ws, size_t ws_size,
                              hipStream_t stream) {
    const float* x    = (const float*)d_in[0];
    const int*   ei   = (const int*)d_in[1];    // [2,E] row-major: row ptr, col ptr
    const float* ew   = (const float*)d_in[2];
    const float* W0   = (const float*)d_in[3];
    const float* Wih  = (const float*)d_in[4];
    const float* Whh  = (const float*)d_in[5];
    const float* bih  = (const float*)d_in[6];
    const float* bhh  = (const float*)d_in[7];
    const float* Wlin = (const float*)d_in[8];
    const float* blin = (const float*)d_in[9];
    float* out = (float*)d_out;

    const int N = in_sizes[0] / FDIM;   // 50000
    const int E = in_sizes[2];          // 800000
    const int* erow = ei;
    const int* ecol = ei + E;

    // workspace carve-up (256B aligned)
    char* ws = (char*)d_ws;
    size_t off = 0;
    auto carve = [&](size_t bytes) {
        void* p = (void*)(ws + off);
        off += (bytes + 255) & ~(size_t)255;
        return p;
    };
    __bf16* Wb_hi = (__bf16*)carve(FDIM * FDIM * sizeof(__bf16));
    __bf16* Wb_lo = (__bf16*)carve(FDIM * FDIM * sizeof(__bf16));
    float*  deg   = (float*)carve((size_t)N * sizeof(float));
    float*  dinv  = (float*)carve((size_t)N * sizeof(float));
    float*  xw    = (float*)carve((size_t)N * FDIM * sizeof(float));
    float*  h     = (float*)carve((size_t)N * FDIM * sizeof(float));
    (void)ws_size;

    // 1. evolve weights -> B-operand layout (bf16 hi/lo)
    gates_kernel<<<64, 256, 0, stream>>>(W0, Wih, Whh, bih, bhh, Wb_hi, Wb_lo);

    // 2. gcn_norm
    deg_init_kernel<<<(N + 255) / 256, 256, 0, stream>>>(deg, N);
    deg_accum_kernel<<<(E + 255) / 256, 256, 0, stream>>>(ecol, ew, deg, E);
    dinv_kernel<<<(N + 255) / 256, 256, 0, stream>>>(deg, dinv, N);

    // 3. xw = x @ W (WMMA)
    int tiles = (N + 15) / 16;
    gemm_xw_kernel<<<(tiles + 3) / 4, 128, 0, stream>>>(x, Wb_hi, Wb_lo, xw, N, tiles);

    // 4. h = self-loop + edge scatter
    hinit_kernel<<<(N * FDIM + 255) / 256, 256, 0, stream>>>(xw, dinv, h, N * FDIM);
    const int sblocks = 2048;
    scatter_kernel<<<sblocks, 256, 0, stream>>>(erow, ecol, ew, dinv, xw, h, E,
                                                sblocks * 256 / 32);

    // 5. out = relu(h) @ Wlin^T + blin
    final_kernel<<<(N + 7) / 8, 256, 0, stream>>>(h, Wlin, blin, out, N);
}
